// StyleEmbedAttention_79559974191187
// MI455X (gfx1250) — matrix operand order repaired and checked by reading, not compiled, not run on hardware
//
#include <hip/hip_runtime.h>
#include <hip/hip_bf16.h>

// ---------------- CDNA5 (gfx1250) fused StyleEmbedAttention ----------------
// q = query @ Wq^T ; k = key @ Wk^T ; v = key @ Wv^T  (all via bf16 WMMA)
// scores = (q_h . k_h)/16 ; softmax over Tk=10 ; out = attn @ v_h
// One block = 8 batch items, 512 threads (16 wave32), everything fused in LDS.
// LDS < 160KB so two workgroups co-reside per WGP. All GEMM phases are
// column-tile-outer with the ct loop forced rolled (#pragma unroll 1) so only
// one 40-VGPR accumulator set + one load pipeline is live at a time (round 2
// unrolled it and spilled to scratch).

#define NUM_UNITS 512
#define NUM_HEADS 8
#define HEAD_DIM 64
#define KEY_DIM 256
#define QUERY_DIM 384
#define TK 10
#define NB 8                 // batch items per block
#define ROWS (NB * TK)       // 80 key/value rows per block
#define THREADS 512

typedef __attribute__((ext_vector_type(16))) __bf16 v16bf;
typedef __attribute__((ext_vector_type(8)))  float  v8f;

// LDS strides (in 16-bit units), padded to dodge bank conflicts, 16B aligned.
#define SA_STRIDE 264   // key rows:   256 + 8
#define SQ_STRIDE 392   // query rows: 384 + 8
#define KV_STRIDE 520   // k/v rows:   512 + 8
#define Q_STRIDE  520   // q rows (bf16): 512 + 8

#define SMEM_BYTES ((ROWS*SA_STRIDE + 16*SQ_STRIDE + ROWS*KV_STRIDE + 16*Q_STRIDE) * 2 \
                    + NB*NUM_HEADS*TK*4)   // = 157,184 bytes (< 160KB)

__device__ __forceinline__ unsigned short f2bf(float f) {
    unsigned u = __float_as_uint(f);
    unsigned r = u + 0x7FFFu + ((u >> 16) & 1u);   // round-to-nearest-even
    return (unsigned short)(r >> 16);
}
__device__ __forceinline__ float bf2f(unsigned short h) {
    return __uint_as_float(((unsigned)h) << 16);
}

union Frag32B { uint4 u[2]; v16bf v; };

// A-fragment (16x32 bf16) from LDS rows (row-major, K contiguous).
// Layout: lane m = l&15; hi = l>>4; VGPR v (0-7): K = hi*16 + 2v.
__device__ __forceinline__ v16bf load_a(const unsigned short* base, int stride,
                                        int lane, int m0, int k0) {
    int m  = m0 + (lane & 15);
    int hi = (lane >> 4) & 1;
    const unsigned short* p = base + m * stride + k0 + hi * 16;
    Frag32B f;
    f.u[0] = *(const uint4*)p;          // ds_load_b128
    f.u[1] = *(const uint4*)(p + 8);    // ds_load_b128
    return f.v;
}

// B-fragment (32x16 bf16) pre-packed in ws as [kt][nt][lane][v] dwords.
__device__ __forceinline__ v16bf load_b(const unsigned* pack, int kt, int nt, int lane) {
    const uint4* p = (const uint4*)(pack + ((((kt << 5) + nt) << 5) + lane) * 8);
    Frag32B f;
    f.u[0] = p[0];                      // global_load_b128
    f.u[1] = p[1];                      // global_load_b128
    return f.v;
}

// Pre-pack W[n][k] (f32, row-major [out,in]) -> bf16 B-fragments B[k][n].
// Dword layout: [15:0] = B[k][n], [31:16] = B[k+1][n] with
// n = nt*16 + (lane&15), k = kt*32 + (lane>>4)*16 + 2v.
__global__ void pack_weights_kernel(const float* __restrict__ W, int K,
                                    unsigned* __restrict__ out, int KT) {
    int idx = blockIdx.x * blockDim.x + threadIdx.x;
    int total = KT * 32 * 32 * 8;
    if (idx >= total) return;
    int v    = idx & 7;
    int lane = (idx >> 3) & 31;
    int nt   = (idx >> 8) & 31;
    int kt   = idx >> 13;
    int n = nt * 16 + (lane & 15);
    int k = kt * 32 + ((lane >> 4) & 1) * 16 + 2 * v;
    unsigned short b0 = f2bf(W[(size_t)n * K + k]);
    unsigned short b1 = f2bf(W[(size_t)n * K + k + 1]);
    out[idx] = (unsigned)b0 | ((unsigned)b1 << 16);
}

__global__ __launch_bounds__(THREADS)
void style_attn_fused_kernel(const float* __restrict__ query,
                             const float* __restrict__ key_soft,
                             const unsigned* __restrict__ Bq,
                             const unsigned* __restrict__ Bk,
                             const unsigned* __restrict__ Bv,
                             float* __restrict__ out) {
    extern __shared__ char smem[];
    unsigned short* sA  = (unsigned short*)smem;            // [80][SA_STRIDE] bf16 key rows
    unsigned short* sAq = sA + ROWS * SA_STRIDE;            // [16][SQ_STRIDE] bf16 query rows
    unsigned short* sKV = sAq + 16 * SQ_STRIDE;             // [80][KV_STRIDE] bf16 k then v
    unsigned short* sQ  = sKV + ROWS * KV_STRIDE;           // [16][Q_STRIDE]  bf16 q
    float* sAttn = (float*)(sQ + 16 * Q_STRIDE);            // [8][8][10] scores->attn

    const int tid  = threadIdx.x;
    const int wave = tid >> 5;
    const int lane = tid & 31;
    const int n0   = blockIdx.x * NB;
    const int c0   = wave * 32;       // this wave's 32-column strip of U
    const int nt0  = wave * 2;
    const int nn   = lane & 15;
    const int hi   = (lane >> 4) & 1;

    // ---- stage key rows (80 x 256 f32 -> bf16 LDS), fully coalesced b128 ----
    {
        const float4* kb4 = (const float4*)(key_soft + (size_t)n0 * TK * KEY_DIM);
        for (int t = tid; t < ROWS * KEY_DIM / 4; t += THREADS) {
            float4 f = kb4[t];                               // global_load_b128
            int flat = t * 4;
            int row = flat >> 8;                             // /256
            int k   = flat & 255;
            unsigned p0 = (unsigned)f2bf(f.x) | ((unsigned)f2bf(f.y) << 16);
            unsigned p1 = (unsigned)f2bf(f.z) | ((unsigned)f2bf(f.w) << 16);
            *(uint2*)(sA + row * SA_STRIDE + k) = make_uint2(p0, p1);  // ds_store_b64
        }
    }
    // ---- stage query rows (8 real + 8 zero-padded, 16 x 384) ----
    {
        const float* qb = query + (size_t)n0 * QUERY_DIM;
        for (int t = tid; t < 16 * QUERY_DIM / 4; t += THREADS) {
            int flat = t * 4;
            int row = flat / QUERY_DIM;
            int k   = flat - row * QUERY_DIM;
            float4 f = make_float4(0.f, 0.f, 0.f, 0.f);
            if (row < NB) f = *(const float4*)(qb + (size_t)row * QUERY_DIM + k);
            unsigned p0 = (unsigned)f2bf(f.x) | ((unsigned)f2bf(f.y) << 16);
            unsigned p1 = (unsigned)f2bf(f.z) | ((unsigned)f2bf(f.w) << 16);
            *(uint2*)(sAq + row * SQ_STRIDE + k) = make_uint2(p0, p1);
        }
    }
    __syncthreads();

    // ---- Q projection: [16 x 384] x [384 x 512] -> sQ (bf16) ----
    // ct loop forced rolled: one 8-VGPR accumulator + one load pipeline live.
#pragma unroll 1
    for (int ct = 0; ct < 2; ++ct) {
        v8f acc = {};
        for (int kt = 0; kt < QUERY_DIM / 32; ++kt) {
            v16bf a = load_a(sAq, SQ_STRIDE, lane, 0, kt * 32);
            v16bf b = load_b(Bq, kt, nt0 + ct, lane);
            acc = __builtin_amdgcn_wmma_f32_16x16x32_bf16(false, a, false, b,
                                                          (short)0, acc, false, false);
        }
#pragma unroll
        for (int r = 0; r < 8; ++r) {
            int row = hi * 8 + r;
            sQ[row * Q_STRIDE + c0 + ct * 16 + nn] = f2bf(acc[r]);
        }
    }

    // ---- K projection: [80 x 256] x [256 x 512] -> sKV (bf16) ----
    // column-tile-outer, rolled: 5 live accumulators (40 VGPRs)
#pragma unroll 1
    for (int ct = 0; ct < 2; ++ct) {
        v8f acc[5];
#pragma unroll
        for (int rt = 0; rt < 5; ++rt) acc[rt] = (v8f){};
        for (int kt = 0; kt < KEY_DIM / 32; ++kt) {
            v16bf b = load_b(Bk, kt, nt0 + ct, lane);
#pragma unroll
            for (int rt = 0; rt < 5; ++rt) {
                v16bf a = load_a(sA, SA_STRIDE, lane, rt * 16, kt * 32);
                acc[rt] = __builtin_amdgcn_wmma_f32_16x16x32_bf16(false, a, false, b,
                                                    (short)0, acc[rt], false, false);
            }
        }
#pragma unroll
        for (int rt = 0; rt < 5; ++rt)
#pragma unroll
            for (int r = 0; r < 8; ++r) {
                int row = rt * 16 + hi * 8 + r;
                sKV[row * KV_STRIDE + c0 + ct * 16 + nn] = f2bf(acc[rt][r]);
            }
    }
    __syncthreads();

    // ---- scores: 8n x 8h x 10t, each a 64-dot over head dim, scale 1/sqrt(256) ----
    for (int s = tid; s < NB * NUM_HEADS * TK; s += THREADS) {
        int t = s % TK;
        int nh = s / TK;
        int h = nh & 7, n = nh >> 3;
        const unsigned short* qp = sQ + n * Q_STRIDE + h * HEAD_DIM;
        const unsigned short* kp = sKV + (n * TK + t) * KV_STRIDE + h * HEAD_DIM;
        float a = 0.f;
#pragma unroll 16
        for (int d = 0; d < HEAD_DIM; ++d) a += bf2f(qp[d]) * bf2f(kp[d]);
        sAttn[s] = a * 0.0625f;
    }
    __syncthreads();
    // ---- softmax over Tk for each (n,h) ----
    if (tid < NB * NUM_HEADS) {
        float* sc = sAttn + tid * TK;
        float m = sc[0];
#pragma unroll
        for (int t = 1; t < TK; ++t) m = fmaxf(m, sc[t]);
        float sum = 0.f;
#pragma unroll
        for (int t = 0; t < TK; ++t) { float e = __expf(sc[t] - m); sc[t] = e; sum += e; }
        float inv = 1.f / sum;
#pragma unroll
        for (int t = 0; t < TK; ++t) sc[t] *= inv;
    }
    __syncthreads();

    // ---- V projection (overwrites sKV) ----
#pragma unroll 1
    for (int ct = 0; ct < 2; ++ct) {
        v8f acc[5];
#pragma unroll
        for (int rt = 0; rt < 5; ++rt) acc[rt] = (v8f){};
        for (int kt = 0; kt < KEY_DIM / 32; ++kt) {
            v16bf b = load_b(Bv, kt, nt0 + ct, lane);
#pragma unroll
            for (int rt = 0; rt < 5; ++rt) {
                v16bf a = load_a(sA, SA_STRIDE, lane, rt * 16, kt * 32);
                acc[rt] = __builtin_amdgcn_wmma_f32_16x16x32_bf16(false, a, false, b,
                                                    (short)0, acc[rt], false, false);
            }
        }
#pragma unroll
        for (int rt = 0; rt < 5; ++rt)
#pragma unroll
            for (int r = 0; r < 8; ++r) {
                int row = rt * 16 + hi * 8 + r;
                sKV[row * KV_STRIDE + c0 + ct * 16 + nn] = f2bf(acc[rt][r]);
            }
    }
    __syncthreads();

    // ---- epilogue: out[n][c] = sum_t attn[n][h(c)][t] * v[n*10+t][c], coalesced ----
    float* outp = out + (size_t)n0 * NUM_UNITS;
    for (int i = tid; i < NB * NUM_UNITS; i += THREADS) {
        int n = i >> 9;
        int c = i & (NUM_UNITS - 1);
        int h = c >> 6;
        const float* at = sAttn + (n * NUM_HEADS + h) * TK;
        float a = 0.f;
#pragma unroll
        for (int t = 0; t < TK; ++t)
            a += at[t] * bf2f(sKV[(n * TK + t) * KV_STRIDE + c]);
        outp[i] = a;
    }
}

extern "C" void kernel_launch(void* const* d_in, const int* in_sizes, int n_in,
                              void* d_out, int out_size, void* d_ws, size_t ws_size,
                              hipStream_t stream) {
    const float* query    = (const float*)d_in[0];  // [32768, 1, 384]
    const float* key_soft = (const float*)d_in[1];  // [32768, 10, 256]
    const float* W_query  = (const float*)d_in[2];  // [512, 384]
    const float* W_key    = (const float*)d_in[3];  // [512, 256]
    const float* W_value  = (const float*)d_in[4];  // [512, 256]
    float* out = (float*)d_out;                     // [32768, 1, 512]

    const int KT_Q = QUERY_DIM / 32;                // 12
    const int KT_K = KEY_DIM / 32;                  // 8
    const int SZ_Q = KT_Q * 32 * 32 * 8;            // 98304 dwords
    const int SZ_K = KT_K * 32 * 32 * 8;            // 65536 dwords
    unsigned* Bq = (unsigned*)d_ws;                 // ws usage: ~0.9 MB total
    unsigned* Bk = Bq + SZ_Q;
    unsigned* Bv = Bk + SZ_K;

    pack_weights_kernel<<<(SZ_Q + 255) / 256, 256, 0, stream>>>(W_query, QUERY_DIM, Bq, KT_Q);
    pack_weights_kernel<<<(SZ_K + 255) / 256, 256, 0, stream>>>(W_key,   KEY_DIM,   Bk, KT_K);
    pack_weights_kernel<<<(SZ_K + 255) / 256, 256, 0, stream>>>(W_value, KEY_DIM,   Bv, KT_K);

    const int N = 32768;
    style_attn_fused_kernel<<<N / NB, THREADS, SMEM_BYTES, stream>>>(
        query, key_soft, Bq, Bk, Bv, out);
}
